// PlainRNN_44762149159378
// MI455X (gfx1250) — matrix operand order repaired and checked by reading, not compiled, not run on hardware
//
#include <hip/hip_runtime.h>
#include <math.h>

// ---------------------------------------------------------------------------
// PlainRNN on MI455X (gfx1250): WMMA f16->f32 GEMMs + persistent-WG scan.
// Round 3: copy-free ping-pong double buffering (2x unrolled K-loop).
// ---------------------------------------------------------------------------

typedef __attribute__((ext_vector_type(16))) _Float16 v16h;
typedef __attribute__((ext_vector_type(8)))  _Float16 v8h;
typedef __attribute__((ext_vector_type(8)))  float    v8f;

#define IN_DIM 128
#define HID    512
#define BATCH  32
#define SEQ_T  2048
#define BT     (BATCH * SEQ_T)          // 65536 rows
#define INV_TAU_X 0.005f                // 1/200

// ---- WMMA helpers ---------------------------------------------------------

__device__ __forceinline__ v8f wmma_f32_16x16x32(v16h a, v16h b, v8f c) {
  // (neg_a, A, neg_b, B, c_mod, C, reuse_a, reuse_b)
  return __builtin_amdgcn_wmma_f32_16x16x32_f16(false, a, false, b, (short)0, c,
                                                false, false);
}

// A-fragment (16x32 f16) per documented layout:
//   lanes 0-15  : row M=lane,    halves = K[0..7]   and K[16..23]
//   lanes 16-31 : row M=lane-16, halves = K[8..15]  and K[24..31]
// p points at row base + k0 + (lane&16 ? 8 : 0); chunks at +0 and +16 halves.
__device__ __forceinline__ v16h load_a_frag(const _Float16* __restrict__ p) {
  v8h lo = *(const v8h*)p;
  v8h hi = *(const v8h*)(p + 16);
  return __builtin_shufflevector(lo, hi, 0, 1, 2, 3, 4, 5, 6, 7,
                                 8, 9, 10, 11, 12, 13, 14, 15);
}

// ---- elementwise f32 -> f16 ----------------------------------------------

__global__ __launch_bounds__(256) void cvt_f32_f16(const float* __restrict__ src,
                                                   _Float16* __restrict__ dst,
                                                   int n) {
  int i = blockIdx.x * 256 + threadIdx.x;
  if (i < n) dst[i] = (_Float16)src[i];
}

// ---- weight packing into WMMA B-fragment tile layout ----------------------
// B is KxN (f16 logical). Tile = 32(K) x 16(N). Packed element order:
//   dst[((nt*Kt + kt)*32 + lane)*16 + j],  lane = ((k_local/8)&1)*16 + n_local,
//   j = (k_local/16)*8 + k_local % 8.  Each lane's 16 halves are contiguous (32 B) so the
//   GEMM loads one v16h per lane per k-step.
// src element (k,n) fetched as src[k*sK + n*sN] (sK/sN allow transpose for W^T).

__global__ __launch_bounds__(256) void pack_b(const float* __restrict__ src,
                                              _Float16* __restrict__ dst,
                                              int K, int N, int sK, int sN) {
  int idx = blockIdx.x * 256 + threadIdx.x;
  if (idx >= K * N) return;
  int j    = idx & 15;
  int lane = (idx >> 4) & 31;
  int tile = idx >> 9;
  int Kt   = K >> 5;
  int kt   = tile % Kt;
  int nt   = tile / Kt;
  int k = kt * 32 + ((j & 8) ? 16 : 0) + (lane >> 4) * 8 + (j & 7);
  int n = nt * 16 + (lane & 15);
  dst[idx] = (_Float16)src[(size_t)k * sK + (size_t)n * sN];
}

// ---- batched GEMM: C = tanh(A @ B + bias) ---------------------------------
// A: MxK f16 row-major. Bp: packed fragments. Each wave computes a
// 2 M-tile x 4 N-tile block (8 WMMA accumulators = 64 VGPRs). K-loop is
// unrolled by 2 with two named fragment register sets (copy-free ping-pong):
// while the 8 WMMAs of set S execute, loads for the iteration after next
// refill set S directly -- no register rotation movs, partial loadcnt waits.
// Requires Kt even (Kt = 4 or 16 here).
// Block = 256 threads = 8 waves -> 256 rows/block. grid = (N/64, M/256).

__device__ __forceinline__ void load_set(v16h a[2], v16h b[4],
                                         const _Float16* __restrict__ ap0,
                                         const _Float16* __restrict__ ap1,
                                         const v16h* __restrict__ bp,
                                         size_t bs, int koff) {
  a[0] = load_a_frag(ap0 + koff);
  a[1] = load_a_frag(ap1 + koff);
#pragma unroll
  for (int j = 0; j < 4; ++j) b[j] = bp[koff + j * bs];
}

__device__ __forceinline__ void compute_set(v8f acc[8], const v16h a[2],
                                            const v16h b[4]) {
#pragma unroll
  for (int mt = 0; mt < 2; ++mt)
#pragma unroll
    for (int j = 0; j < 4; ++j)
      acc[mt * 4 + j] = wmma_f32_16x16x32(a[mt], b[j], acc[mt * 4 + j]);
}

__global__ __launch_bounds__(256)
void gemm_bias_tanh(const _Float16* __restrict__ A,
                    const _Float16* __restrict__ Bp,
                    const float* __restrict__ bias,
                    _Float16* __restrict__ C16,
                    float* __restrict__ C32,
                    int M, int K, int N) {
  const int lane   = threadIdx.x & 31;
  const int wave   = threadIdx.x >> 5;
  const int Kt     = K >> 5;
  const int tileM0 = (blockIdx.y * 8 + wave) * 2;
  const int tileN0 = blockIdx.x * 4;

  const _Float16* ap0 =
      A + (size_t)(tileM0 * 16 + (lane & 15)) * K + ((lane & 16) ? 8 : 0);
  const _Float16* ap1 = ap0 + (size_t)16 * K;
  const v16h* bp = (const v16h*)Bp + (size_t)tileN0 * Kt * 32 + lane;
  const size_t bs = (size_t)Kt * 32;  // v16h units between N tiles

  v8f acc[8] = {};
  v16h aS0[2], bS0[4], aS1[2], bS1[4];
  load_set(aS0, bS0, ap0, ap1, bp, bs, 0);    // kt
  load_set(aS1, bS1, ap0, ap1, bp, bs, 32);   // kt+1

  for (int kt = 0; kt < Kt - 2; kt += 2) {
    compute_set(acc, aS0, bS0);
    load_set(aS0, bS0, ap0, ap1, bp, bs, 64);  // kt+2
    compute_set(acc, aS1, bS1);
    load_set(aS1, bS1, ap0, ap1, bp, bs, 96);  // kt+3
    ap0 += 64;
    ap1 += 64;
    bp  += 64;
  }
  compute_set(acc, aS0, bS0);
  compute_set(acc, aS1, bS1);

#pragma unroll
  for (int mt = 0; mt < 2; ++mt) {
    const int mbase = (tileM0 + mt) * 16 + ((lane & 16) ? 8 : 0);
#pragma unroll
    for (int j = 0; j < 4; ++j) {
      const int ncol = (tileN0 + j) * 16 + (lane & 15);
      const float bv = bias[ncol];
#pragma unroll
      for (int r = 0; r < 8; ++r) {
        float v = tanhf(acc[mt * 4 + j][r] + bv);
        size_t off = (size_t)(mbase + r) * N + ncol;
        if (C16) C16[off] = (_Float16)v;
        if (C32) C32[off] = v;
      }
    }
  }
}

// ---- sequential leaky-RNN scan -------------------------------------------
// Single persistent workgroup (1024 thr = 32 waves). State x (32x512 f32) and
// tanh(x) staging (f16) live in LDS (96 KB of 320 KB WGP LDS). Per step:
//   teacher-force (t%5==0 && t>0), a=tanh(x), y = a @ W^T (WMMA), leaky
//   update, store state (f16) for the batched decoder pass.
// Each wave owns one 16-col tile (tn=wave) across both 16-row tiles. The
// L2-resident B fragment stream is ping-pong double-buffered (copy-free
// under full unroll); A fragments come from LDS (ds_load_b128).

__global__ __launch_bounds__(1024)
void scan_kernel(const _Float16* __restrict__ xseq,   // [B, T, HID] f16
                 const _Float16* __restrict__ WTp,    // packed W^T
                 _Float16* __restrict__ st,           // [B, T, HID] f16 out
                 int T) {
  extern __shared__ char smem[];
  float*    x   = (float*)smem;                        // 32*512 f32 = 64 KB
  _Float16* a16 = (_Float16*)(smem + BATCH * HID * 4); // 32*512 f16 = 32 KB

  const int tid  = threadIdx.x;
  const int lane = tid & 31;
  const int wave = tid >> 5;

  // x0 = x_seq[:, 0, :]
  for (int i = tid; i < BATCH * HID; i += 1024) {
    int b = i >> 9, h = i & 511;
    x[i] = (float)xseq[(size_t)b * T * HID + h];
  }
  __syncthreads();

  for (int t = 0; t < T; ++t) {
    const bool tf = (t > 0) && (t % 5 == 0);
    // stage 1: teacher forcing + a = tanh(x)
    for (int i = tid; i < BATCH * HID; i += 1024) {
      int b = i >> 9, h = i & 511;
      float xv = x[i];
      if (tf) {
        xv = (float)xseq[((size_t)b * T + t) * HID + h];
        x[i] = xv;
      }
      a16[i] = (_Float16)tanhf(xv);
    }
    __syncthreads();

    // stage 2: y = a @ W^T ; x += (y - x)/TAU_X ; store state
    {
      const int tn = wave;  // 32 waves cover all 32 N tiles
      const _Float16* ap0 = a16 + (lane & 15) * HID + ((lane & 16) ? 8 : 0);
      const _Float16* ap1 = ap0 + 16 * HID;
      const v16h* bp = (const v16h*)WTp + (size_t)tn * 16 * 32 + lane; // Kt=16
      v8f acc0 = {}, acc1 = {};
      v16h bfA = bp[0];
      v16h bfB = bp[32];
#pragma unroll
      for (int kt = 0; kt < 16; kt += 2) {
        v16h af0 = load_a_frag(ap0 + kt * 32);
        v16h af1 = load_a_frag(ap1 + kt * 32);
        acc0 = wmma_f32_16x16x32(af0, bfA, acc0);
        acc1 = wmma_f32_16x16x32(af1, bfA, acc1);
        if (kt + 2 < 16) bfA = bp[(kt + 2) * 32];
        af0 = load_a_frag(ap0 + (kt + 1) * 32);
        af1 = load_a_frag(ap1 + (kt + 1) * 32);
        acc0 = wmma_f32_16x16x32(af0, bfB, acc0);
        acc1 = wmma_f32_16x16x32(af1, bfB, acc1);
        if (kt + 3 < 16) bfB = bp[(kt + 3) * 32];
      }
      const int n  = tn * 16 + (lane & 15);
      const int mb = (lane & 16) ? 8 : 0;
#pragma unroll
      for (int r = 0; r < 8; ++r) {
        int m0 = mb + r;           // tile tm=0 rows 0..15
        int m1 = 16 + mb + r;      // tile tm=1 rows 16..31
        int i0 = m0 * HID + n;
        int i1 = m1 * HID + n;
        float xn0 = x[i0] + (acc0[r] - x[i0]) * INV_TAU_X;
        float xn1 = x[i1] + (acc1[r] - x[i1]) * INV_TAU_X;
        x[i0] = xn0;
        x[i1] = xn1;
        st[((size_t)m0 * T + t) * HID + n] = (_Float16)xn0;
        st[((size_t)m1 * T + t) * HID + n] = (_Float16)xn1;
      }
    }
    __syncthreads();
  }
}

// ---------------------------------------------------------------------------

extern "C" void kernel_launch(void* const* d_in, const int* in_sizes, int n_in,
                              void* d_out, int out_size, void* d_ws, size_t ws_size,
                              hipStream_t stream) {
  (void)in_sizes; (void)n_in; (void)out_size; (void)ws_size;

  const float* in_seq = (const float*)d_in[0];
  const float* We1 = (const float*)d_in[1];
  const float* be1 = (const float*)d_in[2];
  const float* We2 = (const float*)d_in[3];
  const float* be2 = (const float*)d_in[4];
  const float* We3 = (const float*)d_in[5];
  const float* be3 = (const float*)d_in[6];
  const float* Wd1 = (const float*)d_in[7];
  const float* bd1 = (const float*)d_in[8];
  const float* Wd2 = (const float*)d_in[9];
  const float* bd2 = (const float*)d_in[10];
  const float* W   = (const float*)d_in[11];

  float* out_pred  = (float*)d_out;                         // [B,T,IN_DIM]
  float* out_recon = out_pred + (size_t)BT * IN_DIM;        // [B,T,IN_DIM]

  // workspace carve-out (reused buffers; ~220 MB total)
  char* ws = (char*)d_ws;
  size_t off = 0;
  auto take = [&](size_t bytes) -> char* {
    char* p = ws + off;
    off += (bytes + 255) & ~(size_t)255;
    return p;
  };
  _Float16* pWe1 = (_Float16*)take((size_t)IN_DIM * HID * 2);
  _Float16* pWe2 = (_Float16*)take((size_t)HID * HID * 2);
  _Float16* pWe3 = (_Float16*)take((size_t)HID * HID * 2);
  _Float16* pWd1 = (_Float16*)take((size_t)HID * HID * 2);
  _Float16* pWd2 = (_Float16*)take((size_t)HID * IN_DIM * 2);
  _Float16* pWT  = (_Float16*)take((size_t)HID * HID * 2);
  _Float16* a0   = (_Float16*)take((size_t)BT * IN_DIM * 2);
  _Float16* a1   = (_Float16*)take((size_t)BT * HID * 2);
  _Float16* a2   = (_Float16*)take((size_t)BT * HID * 2);
  _Float16* xseq = (_Float16*)take((size_t)BT * HID * 2);

  // --- weight packing + input conversion ---
  pack_b<<<(IN_DIM * HID + 255) / 256, 256, 0, stream>>>(We1, pWe1, IN_DIM, HID, HID, 1);
  pack_b<<<(HID * HID + 255) / 256, 256, 0, stream>>>(We2, pWe2, HID, HID, HID, 1);
  pack_b<<<(HID * HID + 255) / 256, 256, 0, stream>>>(We3, pWe3, HID, HID, HID, 1);
  pack_b<<<(HID * HID + 255) / 256, 256, 0, stream>>>(Wd1, pWd1, HID, HID, HID, 1);
  pack_b<<<(HID * IN_DIM + 255) / 256, 256, 0, stream>>>(Wd2, pWd2, HID, IN_DIM, IN_DIM, 1);
  // W^T: element (k,n) = W[n][k]  -> strides (1, HID)
  pack_b<<<(HID * HID + 255) / 256, 256, 0, stream>>>(W, pWT, HID, HID, 1, HID);
  cvt_f32_f16<<<((size_t)BT * IN_DIM + 255) / 256, 256, 0, stream>>>(in_seq, a0, BT * IN_DIM);

  const dim3 blk(256);
  const dim3 gridH(HID / 64, BT / 256);     // N=512 GEMMs
  const dim3 gridI(IN_DIM / 64, BT / 256);  // N=128 GEMMs

  // --- encoder: x_seq = tanh(tanh(tanh(u@We1+b)@We2+b)@We3+b) ---
  gemm_bias_tanh<<<gridH, blk, 0, stream>>>(a0,   pWe1, be1, a1,   nullptr, BT, IN_DIM, HID);
  gemm_bias_tanh<<<gridH, blk, 0, stream>>>(a1,   pWe2, be2, a2,   nullptr, BT, HID, HID);
  gemm_bias_tanh<<<gridH, blk, 0, stream>>>(a2,   pWe3, be3, xseq, nullptr, BT, HID, HID);

  // --- x_recon = decoder(x_seq) ---
  gemm_bias_tanh<<<gridH, blk, 0, stream>>>(xseq, pWd1, bd1, a1, nullptr, BT, HID, HID);
  gemm_bias_tanh<<<gridI, blk, 0, stream>>>(a1,   pWd2, bd2, nullptr, out_recon, BT, HID, IN_DIM);

  // --- sequential scan: states -> a2 (reused) ---
  (void)hipFuncSetAttribute((const void*)scan_kernel,
                            hipFuncAttributeMaxDynamicSharedMemorySize,
                            BATCH * HID * 4 + BATCH * HID * 2);
  scan_kernel<<<1, 1024, BATCH * HID * 4 + BATCH * HID * 2, stream>>>(xseq, pWT, a2, SEQ_T);

  // --- x_pred = decoder(states), batched over all B*T states ---
  gemm_bias_tanh<<<gridH, blk, 0, stream>>>(a2, pWd1, bd1, a1, nullptr, BT, HID, HID);
  gemm_bias_tanh<<<gridI, blk, 0, stream>>>(a1, pWd2, bd2, nullptr, out_pred, BT, HID, IN_DIM);
}
